// GATNet_43920335569312
// MI455X (gfx1250) — matrix operand order repaired and checked by reading, not compiled, not run on hardware
//
#include <hip/hip_runtime.h>
#include <math.h>

// ---------------- types for WMMA ----------------
typedef __attribute__((ext_vector_type(16))) __bf16          v16bf;
typedef __attribute__((ext_vector_type(16))) unsigned short  v16u;
typedef __attribute__((ext_vector_type(8)))  float           v8f;
typedef __attribute__((ext_vector_type(4)))  float           v4f;

__device__ __forceinline__ unsigned short f2bf(float f) {
    unsigned int u = __float_as_uint(f);
    u += 0x7FFFu + ((u >> 16) & 1u);          // round-to-nearest-even
    return (unsigned short)(u >> 16);
}

__device__ __forceinline__ void atomicMaxF(float* addr, float v) {
    if (v >= 0.f) atomicMax((int*)addr, __float_as_int(v));
    else          atomicMin((unsigned int*)addr, (unsigned int)__float_as_int(v));
}

__device__ __forceinline__ float lrelu(float a) { return a > 0.f ? a : 0.2f * a; }

// ---------------- elementwise helpers ----------------
__global__ void k_fillf(float* __restrict__ p, long long n, float v) {
    long long t = (long long)blockIdx.x * blockDim.x + threadIdx.x;
    if (t < n) p[t] = v;
}

__global__ void k_cvt_bf16(const float* __restrict__ in, unsigned short* __restrict__ out,
                           long long n) {
    long long t = (long long)blockIdx.x * blockDim.x + threadIdx.x;
    if (t < n) out[t] = f2bf(in[t]);
}

// W [K, Ncols] (row major) -> Wt bf16 [Ncols, K] so B-tile K-reads are contiguous
__global__ void k_transpose_bf16(const float* __restrict__ W, unsigned short* __restrict__ Wt,
                                 int K, int Ncols) {
    int t = blockIdx.x * blockDim.x + threadIdx.x;
    if (t >= K * Ncols) return;
    int k = t / Ncols, c = t % Ncols;
    Wt[(size_t)c * K + k] = f2bf(W[t]);
}

// ---------------- WMMA GEMM: C[M,Ncols] = A[M,K](bf16) * Bt[Ncols,K](bf16)^T ----------------
// All indexing 32-bit (buffers < 2^31 elements). One wave per 16x16 tile.
template <int K>
__global__ void k_gemm_wmma_bf16(const unsigned short* __restrict__ A,
                                 const unsigned short* __restrict__ Bt,
                                 float* __restrict__ C, int M, int ldc) {
    const int lane  = threadIdx.x & 31;
    const int bm    = blockIdx.x;             // 16-row M tile
    const int cn    = blockIdx.y;             // 16-col N tile
    int row = bm * 16 + (lane & 15);
    if (row >= M) row = M - 1;                // load clamp for ragged tail
    const int col   = cn * 16 + (lane & 15);
    const int abase = (lane >> 4) * 8;        // A K-offset group per ISA 16-bit layout
    const int bbase = (lane >> 4) * 8;        // B K-offset group per ISA 16-bit layout

    const unsigned short* ap = A  + row * K;
    const unsigned short* bp = Bt + col * K;

    v8f acc = {0.f, 0.f, 0.f, 0.f, 0.f, 0.f, 0.f, 0.f};
#pragma unroll
    for (int kk = 0; kk < K; kk += 32) {
        union { v16u u; v16bf b; } a, b;
#pragma unroll
        for (int j = 0; j < 8; ++j) {
            a.u[j]     = ap[kk + abase + j];        // K = base+0..7   (16B contiguous)
            a.u[j + 8] = ap[kk + abase + 16 + j];   // K = base+16..23 (16B contiguous)
        }
#pragma unroll
        for (int j = 0; j < 8; ++j) {
            b.u[j]     = bp[kk + bbase + j];        // K = base+0..7
            b.u[j + 8] = bp[kk + bbase + 16 + j];   // K = base+16..23
        }
        acc = __builtin_amdgcn_wmma_f32_16x16x32_bf16(false, a.b, false, b.b,
                                                      (short)0, acc, false, false);
    }
    const int mrow = bm * 16 + (lane >> 4) * 8;
    float* cp = C + mrow * ldc + col;
    if (bm * 16 + 16 <= M) {                  // uniform branch: full tile -> branchless stores
#pragma unroll
        for (int i = 0; i < 8; ++i) cp[i * ldc] = acc[i];
    } else {
#pragma unroll
        for (int i = 0; i < 8; ++i)
            if (mrow + i < M) cp[i * ldc] = acc[i];
    }
}

// ---------------- attention scores s,d = <xl, a_src/a_dst> per (node, head) ----------------
__global__ void k_attn_scores(const float* __restrict__ xl, const float* __restrict__ asrc,
                              const float* __restrict__ adst, float* __restrict__ s,
                              float* __restrict__ d, long long n, int H, int C) {
    long long t = (long long)blockIdx.x * blockDim.x + threadIdx.x;   // t = node*H + h
    if (t >= n * H) return;
    int h = (int)(t % H);
    const v4f* row = (const v4f*)(xl + t * C);
    const v4f* sa  = (const v4f*)(asrc + h * C);
    const v4f* da  = (const v4f*)(adst + h * C);
    float ss = 0.f, dd = 0.f;
    for (int c = 0; c < C / 4; ++c) {
        v4f v = row[c], va = sa[c], vd = da[c];
#pragma unroll
        for (int j = 0; j < 4; ++j) {
            ss = fmaf(v[j], va[j], ss);
            dd = fmaf(v[j], vd[j], dd);
        }
    }
    s[t] = ss; d[t] = dd;
}

__device__ __forceinline__ void edge_ends(const long long* __restrict__ ei, long long E,
                                          long long e, int& si, int& di) {
    if (e < E) { si = (int)ei[e]; di = (int)ei[E + e]; }
    else       { si = di = (int)(e - E); }          // self-loop
}

// ---------------- edge pass 1: segment max of leaky_relu(s[src]+d[dst]) ----------------
__global__ void k_edge_max(const long long* __restrict__ ei, long long E, long long Etot,
                           int H, const float* __restrict__ s, const float* __restrict__ d,
                           float* __restrict__ amax) {
    long long t = (long long)blockIdx.x * blockDim.x + threadIdx.x;
    if (t >= Etot * H) return;
    long long e = t / H; int h = (int)(t % H);
    int si, di; edge_ends(ei, E, e, si, di);
    float a = lrelu(s[(size_t)si * H + h] + d[(size_t)di * H + h]);
    atomicMaxF(&amax[(size_t)di * H + h], a);
}

// ---------------- edge pass 2 (H=8, C=32): one wave per edge, float4 gathers ----------------
__global__ void k_edge_accum_h8(const long long* __restrict__ ei, long long E, long long Etot,
                                const float* __restrict__ s, const float* __restrict__ d,
                                const float* __restrict__ amax, const float* __restrict__ xl,
                                float* __restrict__ num, float* __restrict__ den) {
    long long wave = ((long long)blockIdx.x * blockDim.x + threadIdx.x) >> 5;
    int lane = threadIdx.x & 31;
    if (wave >= Etot) return;
    int si, di; edge_ends(ei, E, wave, si, di);
    const v4f* srow4 = (const v4f*)(xl + (size_t)si * 256);
    float* nrow = num + (size_t)di * 256;
    __builtin_prefetch((const void*)(srow4 + lane), 0, 1);  // global_prefetch_b8 of gathered row
#pragma unroll
    for (int g = 0; g < 2; ++g) {                  // 2 groups x 128 channels
        int h = g * 4 + (lane >> 3);               // head owning channels lane*4..+3 of group
        float a  = lrelu(s[si * 8 + h] + d[di * 8 + h]);
        float ex = __expf(a - amax[di * 8 + h]);
        v4f v = srow4[g * 32 + lane];              // global_load_b128
        int ch = g * 128 + lane * 4;
#pragma unroll
        for (int j = 0; j < 4; ++j)
            atomicAdd(&nrow[ch + j], ex * v[j]);
    }
    if (lane < 8) {                                // one den update per head
        float a  = lrelu(s[si * 8 + lane] + d[di * 8 + lane]);
        float ex = __expf(a - amax[di * 8 + lane]);
        atomicAdd(&den[(size_t)di * 8 + lane], ex);
    }
}

// ---------------- edge pass 2 (H=1, C=32): one wave per edge, lane = channel ----------------
__global__ void k_edge_accum_h1(const long long* __restrict__ ei, long long E, long long Etot,
                                const float* __restrict__ s, const float* __restrict__ d,
                                const float* __restrict__ amax, const float* __restrict__ xl,
                                float* __restrict__ num, float* __restrict__ den) {
    long long wave = ((long long)blockIdx.x * blockDim.x + threadIdx.x) >> 5;
    int lane = threadIdx.x & 31;
    if (wave >= Etot) return;
    int si, di; edge_ends(ei, E, wave, si, di);
    float a  = lrelu(s[si] + d[di]);
    float ex = __expf(a - amax[di]);
    float v  = ex * xl[(size_t)si * 32 + lane];
    atomicAdd(&num[(size_t)di * 32 + lane], v);
    if (lane == 0) atomicAdd(&den[di], ex);
}

// ---------------- node finish: h = elu(num/den + bias) ----------------
__global__ void k_node_finish(const float* __restrict__ num, const float* __restrict__ den,
                              const float* __restrict__ bias, float* __restrict__ out,
                              long long n, int H, int C) {
    long long t = (long long)blockIdx.x * blockDim.x + threadIdx.x;
    long long F = (long long)H * C;
    if (t >= n * F) return;
    long long node = t / F;
    int j = (int)(t % F);
    int h = j / C;
    float v = num[t] / (den[node * H + h] + 1e-16f) + bias[j];
    out[t] = v > 0.f ? v : (__expf(v) - 1.f);
}

// ---------------- gate = Linear(ReLU(Linear(h2))) , one wave per node ----------------
__global__ void k_gate(const float* __restrict__ h2, const float* __restrict__ Wg1,
                       const float* __restrict__ bg1, const float* __restrict__ Wg2,
                       const float* __restrict__ bg2, float* __restrict__ gate, long long n) {
    long long wave = ((long long)blockIdx.x * blockDim.x + threadIdx.x) >> 5;
    int lane = threadIdx.x & 31;
    if (wave >= n) return;
    const float* hr = h2 + wave * 32;
    float acc = bg1[lane];
    for (int k = 0; k < 32; ++k) acc = fmaf(hr[k], Wg1[k * 32 + lane], acc);
    float g = fmaxf(acc, 0.f) * Wg2[lane];
    for (int off = 16; off > 0; off >>= 1) g += __shfl_xor(g, off, 32);
    if (lane == 0) gate[wave] = g + bg2[0];
}

__global__ void k_pool_max(const float* __restrict__ gate, const long long* __restrict__ batch,
                           float* __restrict__ gmax, long long n) {
    long long t = (long long)blockIdx.x * blockDim.x + threadIdx.x;
    if (t >= n) return;
    atomicMaxF(&gmax[(int)batch[t]], gate[t]);
}

__global__ void k_pool_accum(const float* __restrict__ gate, const float* __restrict__ gmax,
                             const long long* __restrict__ batch, const float* __restrict__ h2,
                             float* __restrict__ pooled, float* __restrict__ deng, long long n) {
    long long wave = ((long long)blockIdx.x * blockDim.x + threadIdx.x) >> 5;
    int lane = threadIdx.x & 31;
    if (wave >= n) return;
    int b = (int)batch[wave];
    float ex = __expf(gate[wave] - gmax[b]);
    atomicAdd(&pooled[(size_t)b * 32 + lane], ex * h2[wave * 32 + lane]);
    if (lane == 0) atomicAdd(&deng[b], ex);
}

__global__ void k_final(const float* __restrict__ pooled, const float* __restrict__ deng,
                        const float* __restrict__ Wf, const float* __restrict__ bf,
                        float* __restrict__ out, int G, int C, int NC) {
    int t = blockIdx.x * blockDim.x + threadIdx.x;
    if (t >= G * NC) return;
    int g = t / NC, j = t % NC;
    float inv = 1.f / (deng[g] + 1e-16f);
    float acc = bf[j];
    for (int c = 0; c < C; ++c)
        acc = fmaf(pooled[(size_t)g * C + c] * inv, Wf[c * NC + j], acc);
    out[t] = acc;
}

// ---------------- host launcher ----------------
static inline long long cdiv(long long a, long long b) { return (a + b - 1) / b; }

extern "C" void kernel_launch(void* const* d_in, const int* in_sizes, int n_in,
                              void* d_out, int out_size, void* d_ws, size_t ws_size,
                              hipStream_t stream) {
    const float*     x     = (const float*)d_in[0];
    const long long* ei    = (const long long*)d_in[1];
    const long long* batch = (const long long*)d_in[2];
    const float* W1  = (const float*)d_in[3];
    const float* as1 = (const float*)d_in[4];
    const float* ad1 = (const float*)d_in[5];
    const float* b1  = (const float*)d_in[6];
    const float* W2  = (const float*)d_in[7];
    const float* as2 = (const float*)d_in[8];
    const float* ad2 = (const float*)d_in[9];
    const float* b2  = (const float*)d_in[10];
    const float* Wg1 = (const float*)d_in[11];
    const float* bg1 = (const float*)d_in[12];
    const float* Wg2 = (const float*)d_in[13];
    const float* bg2 = (const float*)d_in[14];
    const float* Wf  = (const float*)d_in[15];
    const float* bfv = (const float*)d_in[16];
    float* out = (float*)d_out;

    const int IN = 128, HID = 32;
    const int HEADS = in_sizes[4] / HID;            // 8
    const int F1    = HEADS * HID;                  // 256
    const long long Nn   = (long long)in_sizes[0] / IN;   // 50000
    const long long E    = (long long)in_sizes[1] / 2;    // 800000
    const long long Etot = E + Nn;                         // +self loops
    const int NC = in_sizes[15] / HID;              // 2
    const int G  = out_size / NC;                   // 128

    // ---- workspace arena ----
    char* w = (char*)d_ws; size_t off = 0;
    auto alloc = [&](size_t bytes) -> void* {
        off = (off + 255) & ~(size_t)255;
        void* p = w + off; off += bytes; return p;
    };
    unsigned short* xb   = (unsigned short*)alloc((size_t)Nn * IN * 2);
    unsigned short* w1t  = (unsigned short*)alloc((size_t)F1 * IN * 2);
    float* xl1   = (float*)alloc((size_t)Nn * F1 * 4);   // later reused as h1
    float* num1  = (float*)alloc((size_t)Nn * F1 * 4);
    float* s1    = (float*)alloc((size_t)Nn * HEADS * 4);
    float* d1    = (float*)alloc((size_t)Nn * HEADS * 4);
    float* amax1 = (float*)alloc((size_t)Nn * HEADS * 4);
    float* den1  = (float*)alloc((size_t)Nn * HEADS * 4);
    unsigned short* h1b = (unsigned short*)alloc((size_t)Nn * F1 * 2);
    unsigned short* w2t = (unsigned short*)alloc((size_t)HID * F1 * 2);
    float* xl2   = (float*)alloc((size_t)Nn * HID * 4);  // later reused as h2
    float* num2  = (float*)alloc((size_t)Nn * HID * 4);
    float* s2    = (float*)alloc((size_t)Nn * 4);
    float* d2    = (float*)alloc((size_t)Nn * 4);
    float* amax2 = (float*)alloc((size_t)Nn * 4);
    float* den2  = (float*)alloc((size_t)Nn * 4);
    float* gate  = (float*)alloc((size_t)Nn * 4);
    float* gmax  = (float*)alloc((size_t)G * 4);
    float* deng  = (float*)alloc((size_t)G * 4);
    float* pooled= (float*)alloc((size_t)G * HID * 4);
    (void)ws_size;

    const int B = 256;
    const float NEG_INF = -INFINITY;

    // ===== Layer 1 =====
    k_cvt_bf16<<<cdiv(Nn * IN, B), B, 0, stream>>>(x, xb, Nn * IN);
    k_transpose_bf16<<<cdiv((long long)IN * F1, B), B, 0, stream>>>(W1, w1t, IN, F1);
    {
        dim3 grid((unsigned)cdiv(Nn, 16), F1 / 16);
        k_gemm_wmma_bf16<128><<<grid, 32, 0, stream>>>(xb, w1t, xl1, (int)Nn, F1);
    }
    k_attn_scores<<<cdiv(Nn * HEADS, B), B, 0, stream>>>(xl1, as1, ad1, s1, d1, Nn, HEADS, HID);
    k_fillf<<<cdiv(Nn * HEADS, B), B, 0, stream>>>(amax1, Nn * HEADS, NEG_INF);
    k_fillf<<<cdiv(Nn * HEADS, B), B, 0, stream>>>(den1, Nn * HEADS, 0.f);
    k_fillf<<<cdiv(Nn * F1, B), B, 0, stream>>>(num1, Nn * F1, 0.f);
    k_edge_max<<<cdiv(Etot * HEADS, B), B, 0, stream>>>(ei, E, Etot, HEADS, s1, d1, amax1);
    k_edge_accum_h8<<<cdiv(Etot * 32, B), B, 0, stream>>>(ei, E, Etot, s1, d1, amax1,
                                                          xl1, num1, den1);
    float* h1 = xl1;   // reuse: xl1 no longer needed
    k_node_finish<<<cdiv(Nn * F1, B), B, 0, stream>>>(num1, den1, b1, h1, Nn, HEADS, HID);

    // ===== Layer 2 =====
    k_cvt_bf16<<<cdiv(Nn * F1, B), B, 0, stream>>>(h1, h1b, Nn * F1);
    k_transpose_bf16<<<cdiv((long long)F1 * HID, B), B, 0, stream>>>(W2, w2t, F1, HID);
    {
        dim3 grid((unsigned)cdiv(Nn, 16), HID / 16);
        k_gemm_wmma_bf16<256><<<grid, 32, 0, stream>>>(h1b, w2t, xl2, (int)Nn, HID);
    }
    k_attn_scores<<<cdiv(Nn, B), B, 0, stream>>>(xl2, as2, ad2, s2, d2, Nn, 1, HID);
    k_fillf<<<cdiv(Nn, B), B, 0, stream>>>(amax2, Nn, NEG_INF);
    k_fillf<<<cdiv(Nn, B), B, 0, stream>>>(den2, Nn, 0.f);
    k_fillf<<<cdiv(Nn * HID, B), B, 0, stream>>>(num2, Nn * HID, 0.f);
    k_edge_max<<<cdiv(Etot, B), B, 0, stream>>>(ei, E, Etot, 1, s2, d2, amax2);
    k_edge_accum_h1<<<cdiv(Etot * 32, B), B, 0, stream>>>(ei, E, Etot, s2, d2, amax2,
                                                          xl2, num2, den2);
    float* h2 = xl2;   // reuse
    k_node_finish<<<cdiv(Nn * HID, B), B, 0, stream>>>(num2, den2, b2, h2, Nn, 1, HID);

    // ===== Global attention pooling + classifier =====
    k_gate<<<cdiv(Nn * 32, B), B, 0, stream>>>(h2, Wg1, bg1, Wg2, bg2, gate, Nn);
    k_fillf<<<1, 256, 0, stream>>>(gmax, G, NEG_INF);
    k_fillf<<<1, 256, 0, stream>>>(deng, G, 0.f);
    k_fillf<<<cdiv((long long)G * HID, B), B, 0, stream>>>(pooled, (long long)G * HID, 0.f);
    k_pool_max<<<cdiv(Nn, B), B, 0, stream>>>(gate, batch, gmax, Nn);
    k_pool_accum<<<cdiv(Nn * 32, B), B, 0, stream>>>(gate, gmax, batch, h2, pooled, deng, Nn);
    k_final<<<cdiv((long long)G * NC, B), B, 0, stream>>>(pooled, deng, Wf, bfv, out, G, HID, NC);
}
